// WindowMSA_26465588478123
// MI455X (gfx1250) — matrix operand order repaired and checked by reading, not compiled, not run on hardware
//
#include <hip/hip_runtime.h>
#include <hip/hip_bf16.h>

// ---------------------------------------------------------------------------
// WindowMSA for MI455X (gfx1250), wave32 + v_wmma_f32_16x16x32_bf16
//   stage 1: qkv = x @ qkv_w^T + qkv_b      -> bf16 ws, layout [s][b][h][n][d]
//   stage 2: per (b,h): softmax(QK^T*scale + relbias) @ V -> bf16 ws [b][n][h*32+d]
//   stage 3: out = attn @ proj_w^T + proj_b -> f32 d_out
// GEMMs: 256x64 block tile, K-step 32, double-buffered LDS, register-staged
// global loads, grid ordered so N varies fastest (x tile + weights stay in L2).
// ---------------------------------------------------------------------------

typedef __attribute__((ext_vector_type(16))) __bf16 v16bf;
typedef __attribute__((ext_vector_type(8)))  __bf16 v8bf;
typedef __attribute__((ext_vector_type(8)))  float  v8f;

union Frag { v16bf v; v8bf h8[2]; };

#define NWIN      2048
#define NTOK      64
#define DIM       512
#define NHEADS    16
#define HEADD     32
#define QKV_COLS  1536
#define M_TOTAL   (NWIN * NTOK)        // 131072
#define SCALE_F   0.17677669529663687f // 32^-0.5

// GEMM tiling
#define KB   32
#define MB   256   // block rows (8 waves x 32 rows)
#define NB   64    // block cols (4 WMMA tiles)
#define NK   (DIM / KB)   // 16 K-steps
#define LDAP 40    // padded LDS row stride (bf16): 80B rows -> 16B aligned, conflict-free
#define LDBP 40

__device__ __forceinline__ v8f wmma_bf16(const v16bf& a, const v16bf& b, v8f c) {
    return __builtin_amdgcn_wmma_f32_16x16x32_bf16(false, a, false, b, (short)0, c,
                                                   false, false);
}

// ---------------------------------------------------------------------------
// Kernel 1: qkv GEMM.  C(131072 x 1536) = X(131072 x 512) * W^T + bias,
// output scattered as bf16 [s][b][h][n][d].
// ---------------------------------------------------------------------------
__global__ __launch_bounds__(256)
void qkv_gemm_kernel(const float* __restrict__ X, const float* __restrict__ W,
                     const float* __restrict__ bias, __bf16* __restrict__ qkv_out)
{
    __shared__ __bf16 As[2][MB * LDAP];
    __shared__ __bf16 Bs[2][NB * LDBP];

    const int tid  = threadIdx.x;
    const int lane = tid & 31;
    const int wave = tid >> 5;
    const int n0   = blockIdx.x * NB;   // N fastest -> x tile reused from L2
    const int m0   = blockIdx.y * MB;
    const int K    = DIM;

    float4 ra[8], rb[2];   // register-staged tiles: loads issue back-to-back

    auto load_tiles = [&](int k0) {
#pragma unroll
        for (int j = 0; j < 8; ++j) {      // A: 256x32 f32 = 2048 float4
            const int lin = tid + j * 256;
            const int r = lin >> 3, g = lin & 7;
            ra[j] = *(const float4*)(X + (size_t)(m0 + r) * K + k0 + g * 4);
        }
#pragma unroll
        for (int j = 0; j < 2; ++j) {      // B: 64x32 f32 = 512 float4
            const int lin = tid + j * 256;
            const int r = lin >> 3, g = lin & 7;
            rb[j] = *(const float4*)(W + (size_t)(n0 + r) * K + k0 + g * 4);
        }
    };
    auto store_tiles = [&](int buf) {
#pragma unroll
        for (int j = 0; j < 8; ++j) {
            const int lin = tid + j * 256;
            const int r = lin >> 3, g = lin & 7;
            __bf16* dst = &As[buf][r * LDAP + g * 4];
            dst[0] = (__bf16)ra[j].x; dst[1] = (__bf16)ra[j].y;
            dst[2] = (__bf16)ra[j].z; dst[3] = (__bf16)ra[j].w;
        }
#pragma unroll
        for (int j = 0; j < 2; ++j) {
            const int lin = tid + j * 256;
            const int r = lin >> 3, g = lin & 7;
            __bf16* dst = &Bs[buf][r * LDBP + g * 4];
            dst[0] = (__bf16)rb[j].x; dst[1] = (__bf16)rb[j].y;
            dst[2] = (__bf16)rb[j].z; dst[3] = (__bf16)rb[j].w;
        }
    };

    v8f acc[4][2] = {};
    const int l15 = lane & 15;
    const int ak  = (lane >> 4) << 3;   // 0 or 8
    const int bk  = (lane >> 4) << 3;   // 0 or 8

    load_tiles(0);
    store_tiles(0);
    __syncthreads();

    for (int kt = 0; kt < NK; ++kt) {
        const int cur = kt & 1;
        if (kt + 1 < NK) load_tiles((kt + 1) * KB);   // overlap with compute

        Frag a[2], bf[4];
#pragma unroll
        for (int m = 0; m < 2; ++m) {
            const int row = (wave << 5) + (m << 4) + l15;
            a[m].h8[0] = *(const v8bf*)&As[cur][row * LDAP + ak];
            a[m].h8[1] = *(const v8bf*)&As[cur][row * LDAP + 16 + ak];
        }
#pragma unroll
        for (int t = 0; t < 4; ++t) {
            const int nr = (t << 4) + l15;
            bf[t].h8[0] = *(const v8bf*)&Bs[cur][nr * LDBP + bk];
            bf[t].h8[1] = *(const v8bf*)&Bs[cur][nr * LDBP + bk + 16];
        }
#pragma unroll
        for (int t = 0; t < 4; ++t)
#pragma unroll
            for (int m = 0; m < 2; ++m)
                acc[t][m] = wmma_bf16(a[m].v, bf[t].v, acc[t][m]);

        if (kt + 1 < NK) store_tiles(cur ^ 1);
        __syncthreads();
    }

    // Store: C layout lane 0-15 -> M=v, lane 16-31 -> M=v+8; col = lane&15 in tile.
    const int rsel = (lane >> 4) << 3;
#pragma unroll
    for (int t = 0; t < 4; ++t) {
        const int col = n0 + (t << 4) + l15;
        const int s = col >> 9, h = (col >> 5) & 15, d = col & 31;
        const float bv = bias[col];
#pragma unroll
        for (int m = 0; m < 2; ++m)
#pragma unroll
            for (int v = 0; v < 8; ++v) {
                const int row  = m0 + (wave << 5) + (m << 4) + v + rsel;
                const int bwin = row >> 6, n = row & 63;
                qkv_out[((((size_t)s * NWIN + bwin) * NHEADS + h) * NTOK + n) * HEADD + d]
                    = (__bf16)(acc[t][m][v] + bv);
            }
    }
}

// ---------------------------------------------------------------------------
// Kernel 2: fused attention per (window b, head h). 128 threads = 4 waves,
// wave w owns query rows [16w, 16w+16).
// ---------------------------------------------------------------------------
__global__ __launch_bounds__(128)
void attn_kernel(const __bf16* __restrict__ qkv, const float* __restrict__ bias_table,
                 const int* __restrict__ rel_index, __bf16* __restrict__ attn_out)
{
    constexpr int LKP = 40;  // K-tile LDS row pad
    constexpr int LVP = 72;  // V^T LDS row pad
    constexpr int LPP = 72;  // P strip LDS row pad
    __shared__ __bf16 Ks[64 * LKP];      // K: [key][d] (B-fragment source)
    __shared__ __bf16 Vt[32 * LVP];      // V transposed: [d][key]
    __shared__ __bf16 Ps[4][16 * LPP];   // per-wave probability strip (transpose buf)

    const int b = blockIdx.x, h = blockIdx.y;
    const int tid = threadIdx.x, lane = tid & 31, wave = tid >> 5;
    const size_t head_sz = (size_t)NTOK * HEADD;
    const __bf16* Qg = qkv + (((size_t)0 * NWIN + b) * NHEADS + h) * head_sz;
    const __bf16* Kg = qkv + (((size_t)1 * NWIN + b) * NHEADS + h) * head_sz;
    const __bf16* Vg = qkv + (((size_t)2 * NWIN + b) * NHEADS + h) * head_sz;

    // Stage K: 64x32 bf16 = 256 x 16B, 2 per thread.
#pragma unroll
    for (int i = 0; i < 2; ++i) {
        const int lin = tid + i * 128;
        const int r = lin >> 2, g = lin & 3;
        *(v8bf*)&Ks[r * LKP + g * 8] = *(const v8bf*)(Kg + r * HEADD + g * 8);
    }
    // Stage V transposed: Vt[d][key].
#pragma unroll
    for (int i = 0; i < 16; ++i) {
        const int lin = tid + i * 128;
        const int n = lin >> 5, d = lin & 31;
        Vt[d * LVP + n] = Vg[lin];
    }
    __syncthreads();

    const int qm  = lane & 15;
    const int ak  = (lane >> 4) << 3;    // 0 or 8
    const int bk  = (lane >> 4) << 3;    // 0 or 8
    const int l15 = lane & 15;

    // Q A-fragment straight from global (ws layout already matches A layout).
    Frag aq;
    aq.h8[0] = *(const v8bf*)(Qg + (wave * 16 + qm) * HEADD + ak);
    aq.h8[1] = *(const v8bf*)(Qg + (wave * 16 + qm) * HEADD + 16 + ak);

    // Preload all 4 K B-fragments, then issue the 4 QK^T WMMAs back-to-back.
    Frag kb[4];
#pragma unroll
    for (int t = 0; t < 4; ++t) {
        const int nr = t * 16 + l15;
        kb[t].h8[0] = *(const v8bf*)&Ks[nr * LKP + bk];
        kb[t].h8[1] = *(const v8bf*)&Ks[nr * LKP + bk + 16];
    }
    v8f s[4] = {};
#pragma unroll
    for (int t = 0; t < 4; ++t) s[t] = wmma_bf16(aq.v, kb[t].v, s[t]);

    // scale + rel-pos bias + row softmax. C layout: lanes<16 hold row v (col=lane),
    // lanes>=16 hold row v+8 — xor-shuffles over {1,2,4,8} reduce each half.
    const int rsel = (lane >> 4) << 3;
#pragma unroll
    for (int v = 0; v < 8; ++v) {
        const int q = wave * 16 + v + rsel;
        float sv[4];
#pragma unroll
        for (int t = 0; t < 4; ++t) {
            const int kk = t * 16 + l15;
            const int idx = rel_index[q * NTOK + kk];
            sv[t] = s[t][v] * SCALE_F + bias_table[idx * NHEADS + h];
        }
        float mx = fmaxf(fmaxf(sv[0], sv[1]), fmaxf(sv[2], sv[3]));
#pragma unroll
        for (int o = 1; o < 16; o <<= 1) mx = fmaxf(mx, __shfl_xor(mx, o, 32));
        float sum = 0.f;
#pragma unroll
        for (int t = 0; t < 4; ++t) { sv[t] = __expf(sv[t] - mx); sum += sv[t]; }
#pragma unroll
        for (int o = 1; o < 16; o <<= 1) sum += __shfl_xor(sum, o, 32);
        const float r = 1.0f / sum;
#pragma unroll
        for (int t = 0; t < 4; ++t)
            Ps[wave][(v + rsel) * LPP + t * 16 + l15] = (__bf16)(sv[t] * r);
    }
    __syncthreads();   // P strip visible before fragment re-read (LDS transpose)

    // O(16x32) = P(16x64) * V(64x32): preload fragments, then 4 WMMAs.
    Frag ap[2], bv[2][2];
#pragma unroll
    for (int kc = 0; kc < 2; ++kc) {
        ap[kc].h8[0] = *(const v8bf*)&Ps[wave][qm * LPP + kc * 32 + ak];
        ap[kc].h8[1] = *(const v8bf*)&Ps[wave][qm * LPP + kc * 32 + 16 + ak];
#pragma unroll
        for (int nt = 0; nt < 2; ++nt) {
            const int nr = nt * 16 + l15;
            bv[nt][kc].h8[0] = *(const v8bf*)&Vt[nr * LVP + kc * 32 + bk];
            bv[nt][kc].h8[1] = *(const v8bf*)&Vt[nr * LVP + kc * 32 + bk + 16];
        }
    }
    v8f o[2] = {};
#pragma unroll
    for (int nt = 0; nt < 2; ++nt)
#pragma unroll
        for (int kc = 0; kc < 2; ++kc)
            o[nt] = wmma_bf16(ap[kc].v, bv[nt][kc].v, o[nt]);

#pragma unroll
    for (int nt = 0; nt < 2; ++nt)
#pragma unroll
        for (int v = 0; v < 8; ++v) {
            const int n = wave * 16 + v + rsel;
            const int d = h * HEADD + nt * 16 + l15;
            attn_out[((size_t)b * NTOK + n) * DIM + d] = (__bf16)o[nt][v];
        }
}

// ---------------------------------------------------------------------------
// Kernel 3: proj GEMM.  out(131072 x 512) = A_bf16 * proj_w^T + proj_b (f32).
// ---------------------------------------------------------------------------
__global__ __launch_bounds__(256)
void proj_gemm_kernel(const __bf16* __restrict__ A, const float* __restrict__ W,
                      const float* __restrict__ bias, float* __restrict__ out)
{
    __shared__ __bf16 As[2][MB * LDAP];
    __shared__ __bf16 Bs[2][NB * LDBP];

    const int tid  = threadIdx.x;
    const int lane = tid & 31;
    const int wave = tid >> 5;
    const int n0   = blockIdx.x * NB;   // N fastest -> A tile reused from L2
    const int m0   = blockIdx.y * MB;
    const int K    = DIM;

    v8bf  ra[4];   // A: 256x32 bf16 = 1024 x 16B, 4 per thread
    float4 rb[2];  // B: 64x32 f32   = 512 float4, 2 per thread

    auto load_tiles = [&](int k0) {
#pragma unroll
        for (int j = 0; j < 4; ++j) {
            const int lin = tid + j * 256;
            const int r = lin >> 2, g = lin & 3;
            ra[j] = *(const v8bf*)(A + (size_t)(m0 + r) * K + k0 + g * 8);
        }
#pragma unroll
        for (int j = 0; j < 2; ++j) {
            const int lin = tid + j * 256;
            const int r = lin >> 3, g = lin & 7;
            rb[j] = *(const float4*)(W + (size_t)(n0 + r) * K + k0 + g * 4);
        }
    };
    auto store_tiles = [&](int buf) {
#pragma unroll
        for (int j = 0; j < 4; ++j) {
            const int lin = tid + j * 256;
            const int r = lin >> 2, g = lin & 3;
            *(v8bf*)&As[buf][r * LDAP + g * 8] = ra[j];
        }
#pragma unroll
        for (int j = 0; j < 2; ++j) {
            const int lin = tid + j * 256;
            const int r = lin >> 3, g = lin & 7;
            __bf16* dst = &Bs[buf][r * LDBP + g * 4];
            dst[0] = (__bf16)rb[j].x; dst[1] = (__bf16)rb[j].y;
            dst[2] = (__bf16)rb[j].z; dst[3] = (__bf16)rb[j].w;
        }
    };

    v8f acc[4][2] = {};
    const int l15 = lane & 15;
    const int ak  = (lane >> 4) << 3;
    const int bk  = (lane >> 4) << 3;

    load_tiles(0);
    store_tiles(0);
    __syncthreads();

    for (int kt = 0; kt < NK; ++kt) {
        const int cur = kt & 1;
        if (kt + 1 < NK) load_tiles((kt + 1) * KB);

        Frag a[2], bf[4];
#pragma unroll
        for (int m = 0; m < 2; ++m) {
            const int row = (wave << 5) + (m << 4) + l15;
            a[m].h8[0] = *(const v8bf*)&As[cur][row * LDAP + ak];
            a[m].h8[1] = *(const v8bf*)&As[cur][row * LDAP + 16 + ak];
        }
#pragma unroll
        for (int t = 0; t < 4; ++t) {
            const int nr = (t << 4) + l15;
            bf[t].h8[0] = *(const v8bf*)&Bs[cur][nr * LDBP + bk];
            bf[t].h8[1] = *(const v8bf*)&Bs[cur][nr * LDBP + bk + 16];
        }
#pragma unroll
        for (int t = 0; t < 4; ++t)
#pragma unroll
            for (int m = 0; m < 2; ++m)
                acc[t][m] = wmma_bf16(a[m].v, bf[t].v, acc[t][m]);

        if (kt + 1 < NK) store_tiles(cur ^ 1);
        __syncthreads();
    }

    const int rsel = (lane >> 4) << 3;
#pragma unroll
    for (int t = 0; t < 4; ++t) {
        const int col = n0 + (t << 4) + l15;
        const float bv = bias[col];
#pragma unroll
        for (int m = 0; m < 2; ++m)
#pragma unroll
            for (int v = 0; v < 8; ++v) {
                const int row = m0 + (wave << 5) + (m << 4) + v + rsel;
                out[(size_t)row * DIM + col] = acc[t][m][v] + bv;
            }
    }
}

// ---------------------------------------------------------------------------
extern "C" void kernel_launch(void* const* d_in, const int* in_sizes, int n_in,
                              void* d_out, int out_size, void* d_ws, size_t ws_size,
                              hipStream_t stream) {
    const float* x          = (const float*)d_in[0];
    const float* qkv_w      = (const float*)d_in[1];
    const float* qkv_b      = (const float*)d_in[2];
    const float* proj_w     = (const float*)d_in[3];
    const float* proj_b     = (const float*)d_in[4];
    const float* bias_table = (const float*)d_in[5];
    const int*   rel_index  = (const int*)d_in[6];
    float*       out        = (float*)d_out;

    // Workspace: qkv (bf16, [3][B][H][N][hd]) then attn_out (bf16, [B][N][C]).
    __bf16* qkv_ws  = (__bf16*)d_ws;
    __bf16* attn_ws = qkv_ws + (size_t)3 * NWIN * NHEADS * NTOK * HEADD;

    dim3 g1(QKV_COLS / NB, M_TOTAL / MB);   // 24 x 512, N fastest for L2 reuse
    qkv_gemm_kernel<<<g1, 256, 0, stream>>>(x, qkv_w, qkv_b, qkv_ws);

    dim3 g2(NWIN, NHEADS);                  // 2048 x 16
    attn_kernel<<<g2, 128, 0, stream>>>(qkv_ws, bias_table, rel_index, attn_ws);

    dim3 g3(DIM / NB, M_TOTAL / MB);        // 8 x 512, N fastest for L2 reuse
    proj_gemm_kernel<<<g3, 256, 0, stream>>>(attn_ws, proj_w, proj_b, out);
}
